// TemporalMamba_57638461112808
// MI455X (gfx1250) — matrix operand order repaired and checked by reading, not compiled, not run on hardware
//
#include <hip/hip_runtime.h>
#include <math.h>

// ---------------------------------------------------------------------------
// Types for CDNA5 WMMA (wave32): v_wmma_f32_16x16x32_bf16
// ---------------------------------------------------------------------------
typedef __bf16 bf16_t;
typedef bf16_t v16bf __attribute__((ext_vector_type(16)));
typedef float  v8f   __attribute__((ext_vector_type(8)));
typedef unsigned short u16;

union FragAB { v16bf v; uint4 q[2]; };

__device__ __forceinline__ v8f wmma_bf16(const FragAB& a, const FragAB& b, v8f c) {
    // (neg_a, A, neg_b, B, c_mod, C, reuse_a, reuse_b)
    return __builtin_amdgcn_wmma_f32_16x16x32_bf16(false, a.v, false, b.v,
                                                   (short)0, c, false, false);
}

__device__ __forceinline__ u16 f2bf(float f) {
    unsigned u = __float_as_uint(f);
    unsigned r = u + 0x7FFFu + ((u >> 16) & 1u);
    return (u16)(r >> 16);
}
__device__ __forceinline__ float bf2f(u16 h) {
    return __uint_as_float(((unsigned)h) << 16);
}

// CDNA5 async global->LDS DMA (tracked by ASYNCcnt). 16B per enabled lane.
__device__ __forceinline__ void async_ld_b128(unsigned lds_addr, const void* gptr) {
    asm volatile("global_load_async_to_lds_b128 %0, %1, off"
                 :: "v"(lds_addr), "v"((unsigned long long)(size_t)gptr)
                 : "memory");
}
__device__ __forceinline__ void wait_async_le1() {
    asm volatile("s_wait_asynccnt 0x1" ::: "memory");
}
__device__ __forceinline__ void wait_async_le0() {
    asm volatile("s_wait_asynccnt 0x0" ::: "memory");
}

#define EPS 1e-5f
#define L_SEQ 4096
#define C_CH  256
#define DI    512
#define DS    16
#define DTR   16
#define NDIR  4
#define NT_TOT (NDIR * L_SEQ)   // 16384

// ===========================================================================
// Prep kernels: BN folding + bf16 weight casts
// ===========================================================================
__global__ void k_cast_bf16(const float* __restrict__ src, u16* __restrict__ dst, int n) {
    for (int i = blockIdx.x * blockDim.x + threadIdx.x; i < n; i += gridDim.x * blockDim.x)
        dst[i] = f2bf(src[i]);
}

// conv3_w (128,512,3,3) -> w3p [9][128][512] bf16 scaled by BN; bias b3f[128]
__global__ void k_prep_c3(const float* __restrict__ w, const float* g, const float* b,
                          const float* m, const float* v,
                          u16* __restrict__ w3p, float* __restrict__ b3f) {
    int idx = blockIdx.x * blockDim.x + threadIdx.x;
    if (idx < 9 * 128 * 512) {
        int tap = idx / (128 * 512);
        int rem = idx - tap * 128 * 512;
        int o = rem / 512, i = rem - o * 512;
        float s = g[o] * rsqrtf(v[o] + EPS);
        w3p[(size_t)(tap * 128 + o) * 512 + i] = f2bf(w[(size_t)(o * 512 + i) * 9 + tap] * s);
    }
    if (idx < 128) {
        float s = g[idx] * rsqrtf(v[idx] + EPS);
        b3f[idx] = b[idx] - m[idx] * s;
    }
}

// 1x1 conv weight [O][I] scaled by BN; bias
__global__ void k_prep_c1(const float* __restrict__ w, const float* g, const float* b,
                          const float* m, const float* v, int O, int I,
                          u16* __restrict__ wp, float* __restrict__ bf) {
    int idx = blockIdx.x * blockDim.x + threadIdx.x;
    if (idx < O * I) {
        int o = idx / I;
        float s = g[o] * rsqrtf(v[o] + EPS);
        wp[idx] = f2bf(w[idx] * s);
    }
    if (idx < O) {
        float s = g[idx] * rsqrtf(v[idx] + EPS);
        bf[idx] = b[idx] - m[idx] * s;
    }
}

// dtproj_w (512,16) -> dtpp [512][32] bf16 (upper 16 cols zero)
__global__ void k_prep_dt(const float* __restrict__ w, u16* __restrict__ dtpp) {
    int idx = blockIdx.x * blockDim.x + threadIdx.x;
    if (idx >= 512 * 32) return;
    int o = idx >> 5, j = idx & 31;
    dtpp[idx] = (j < DTR) ? f2bf(w[o * DTR + j]) : (u16)0;
}

// Aneg = -exp(A_log)  [512][16]
__global__ void k_prep_A(const float* __restrict__ alog, float* __restrict__ an) {
    int idx = blockIdx.x * blockDim.x + threadIdx.x;
    if (idx < 512 * 16) an[idx] = -__expf(alog[idx]);
}

// ===========================================================================
// Generic 8-wave WMMA GEMM block:  C[M][N] = W[M][K] * Act[N][K]^T
// Block = 256 threads (8 waves). Per wave: one 16-row M tile, 4 N tiles of 16.
// Activation tiles [64][32] double-buffered in LDS via async global->LDS DMA:
// tile k+1 streams in (ASYNCcnt) while WMMAs consume tile k.
// ===========================================================================
__device__ __forceinline__ void gemm_block_bf16(
    const u16* __restrict__ W, const u16* __restrict__ Act,
    int K, int M, int n0, u16* lds /* [2][64*32] */, v8f acc[4], int m0)
{
    const int tid  = threadIdx.x;
    const int lane = tid & 31;
    const int col  = lane & 15;
    const int kb   = (lane >> 4) * 16;  // B frag K base (halves)
    const int koff = (lane >> 4) * 16;  // A frag K offset (halves)
    const int arow = m0 + (lane & 15);

    v8f z = {0.f, 0.f, 0.f, 0.f, 0.f, 0.f, 0.f, 0.f};
    acc[0] = z; acc[1] = z; acc[2] = z; acc[3] = z;

    const int srow = tid >> 2;          // 0..63
    const int sjj  = (tid & 3) * 8;     // 0,8,16,24
    const u16* srcp = Act + (size_t)(n0 + srow) * K + sjj;

    const unsigned lb0 = (unsigned)(size_t)(lds + srow * 32 + sjj);
    const unsigned lb1 = (unsigned)(size_t)(lds + 64 * 32 + srow * 32 + sjj);

    // prologue: kick off tile 0 into buffer 0
    async_ld_b128(lb0, srcp);

    int p = 0;
    for (int k0 = 0; k0 < K; k0 += 32) {
        const bool hasNext = (k0 + 32 < K);
        if (hasNext) {
            // stream tile k0+32 into the other buffer (overlaps with compute)
            async_ld_b128(p ? lb0 : lb1, srcp + k0 + 32);
            wait_async_le1();   // in-order: guarantees tile k0 has landed
        } else {
            wait_async_le0();
        }
        __syncthreads();

        if (m0 < M) {
            FragAB a;
            const u16* wp = W + (size_t)arow * K + k0 + koff;
            a.q[0] = *reinterpret_cast<const uint4*>(wp);
            a.q[1] = *reinterpret_cast<const uint4*>(wp + 8);
            const u16* lb = lds + (p ? 64 * 32 : 0);
            FragAB b0, b1, b2, b3;
            {
                const u16* bp = lb + (0 * 16 + col) * 32 + kb;
                b0.q[0] = *reinterpret_cast<const uint4*>(bp);
                b0.q[1] = *reinterpret_cast<const uint4*>(bp + 8);
            }
            {
                const u16* bp = lb + (1 * 16 + col) * 32 + kb;
                b1.q[0] = *reinterpret_cast<const uint4*>(bp);
                b1.q[1] = *reinterpret_cast<const uint4*>(bp + 8);
            }
            {
                const u16* bp = lb + (2 * 16 + col) * 32 + kb;
                b2.q[0] = *reinterpret_cast<const uint4*>(bp);
                b2.q[1] = *reinterpret_cast<const uint4*>(bp + 8);
            }
            {
                const u16* bp = lb + (3 * 16 + col) * 32 + kb;
                b3.q[0] = *reinterpret_cast<const uint4*>(bp);
                b3.q[1] = *reinterpret_cast<const uint4*>(bp + 8);
            }
            acc[0] = wmma_bf16(a, b0, acc[0]);
            acc[1] = wmma_bf16(a, b1, acc[1]);
            acc[2] = wmma_bf16(a, b2, acc[2]);
            acc[3] = wmma_bf16(a, b3, acc[3]);
        }
        __syncthreads();
        p ^= 1;
    }
}

#define GEMM_LDS_U16 (2 * 64 * 32)

// ---- GEMM epilogue variants --------------------------------------------

// bf16 transposed output out[n][m] (ld=M), optional bias + relu
__global__ __launch_bounds__(256) void k_gemm_tbf16(
    const u16* __restrict__ W, const u16* __restrict__ Act,
    const float* __restrict__ bias, u16* __restrict__ out, int K, int M, int relu)
{
    __shared__ alignas(16) u16 lds[GEMM_LDS_U16];
    int wave = threadIdx.x >> 5, lane = threadIdx.x & 31;
    int m0 = (blockIdx.y * 8 + wave) * 16;
    int n0 = blockIdx.x * 64;
    v8f acc[4];
    gemm_block_bf16(W, Act, K, M, n0, lds, acc, m0);
    if (m0 >= M) return;
    int mb = m0 + ((lane >> 4) ? 8 : 0);
#pragma unroll
    for (int nt = 0; nt < 4; ++nt) {
        int n = n0 + nt * 16 + (lane & 15);
        alignas(16) u16 pk[8];
#pragma unroll
        for (int r = 0; r < 8; ++r) {
            float vv = acc[nt][r] + (bias ? bias[mb + r] : 0.f);
            if (relu) vv = fmaxf(vv, 0.f);
            pk[r] = f2bf(vv);
        }
        *reinterpret_cast<uint4*>(out + (size_t)n * M + mb) =
            *reinterpret_cast<const uint4*>(pk);
    }
}

// f32 transposed output out[n][m] (ld=M), + bias
__global__ __launch_bounds__(256) void k_gemm_f32(
    const u16* __restrict__ W, const u16* __restrict__ Act,
    const float* __restrict__ bias, float* __restrict__ out, int K, int M)
{
    __shared__ alignas(16) u16 lds[GEMM_LDS_U16];
    int wave = threadIdx.x >> 5, lane = threadIdx.x & 31;
    int m0 = (blockIdx.y * 8 + wave) * 16;
    int n0 = blockIdx.x * 64;
    v8f acc[4];
    gemm_block_bf16(W, Act, K, M, n0, lds, acc, m0);
    if (m0 >= M) return;
    int mb = m0 + ((lane >> 4) ? 8 : 0);
#pragma unroll
    for (int nt = 0; nt < 4; ++nt) {
        int n = n0 + nt * 16 + (lane & 15);
        float4 lo = make_float4(acc[nt][0] + bias[mb + 0], acc[nt][1] + bias[mb + 1],
                                acc[nt][2] + bias[mb + 2], acc[nt][3] + bias[mb + 3]);
        float4 hi = make_float4(acc[nt][4] + bias[mb + 4], acc[nt][5] + bias[mb + 5],
                                acc[nt][6] + bias[mb + 6], acc[nt][7] + bias[mb + 7]);
        float* o = out + (size_t)n * M + mb;
        *reinterpret_cast<float4*>(o)     = lo;
        *reinterpret_cast<float4*>(o + 4) = hi;
    }
}

// in_proj: M=1024. rows<512 -> xi_raw [dir][ch][t] f32 ; rows>=512 -> silu(zg) [n][512] f32
__global__ __launch_bounds__(256) void k_gemm_inproj(
    const u16* __restrict__ W, const u16* __restrict__ Act,
    float* __restrict__ xi_raw, float* __restrict__ szg)
{
    __shared__ alignas(16) u16 lds[GEMM_LDS_U16];
    int wave = threadIdx.x >> 5, lane = threadIdx.x & 31;
    int m0 = (blockIdx.y * 8 + wave) * 16;
    int n0 = blockIdx.x * 64;
    v8f acc[4];
    gemm_block_bf16(W, Act, C_CH, 1024, n0, lds, acc, m0);
    int mb = m0 + ((lane >> 4) ? 8 : 0);
#pragma unroll
    for (int nt = 0; nt < 4; ++nt) {
        int n = n0 + nt * 16 + (lane & 15);
        int dir = n >> 12, t = n & 4095;
        if (m0 < DI) {
#pragma unroll
            for (int r = 0; r < 8; ++r)
                xi_raw[((size_t)dir * DI + mb + r) * L_SEQ + t] = acc[nt][r];
        } else {
            float o[8];
#pragma unroll
            for (int r = 0; r < 8; ++r) {
                float vv = acc[nt][r];
                o[r] = vv / (1.f + __expf(-vv));
            }
            float* dst = szg + (size_t)n * DI + (mb - DI);
            *reinterpret_cast<float4*>(dst)     = make_float4(o[0], o[1], o[2], o[3]);
            *reinterpret_cast<float4*>(dst + 4) = make_float4(o[4], o[5], o[6], o[7]);
        }
    }
}

// xproj: M=48. m-tile 0 -> dt (bf16, padded to 32 cols); tile 1 -> Bm f32; tile 2 -> Cm f32
__global__ __launch_bounds__(256) void k_gemm_xproj(
    const u16* __restrict__ W, const u16* __restrict__ Act,
    u16* __restrict__ dtpad, float* __restrict__ Bm, float* __restrict__ Cm)
{
    __shared__ alignas(16) u16 lds[GEMM_LDS_U16];
    int wave = threadIdx.x >> 5, lane = threadIdx.x & 31;
    int m0 = wave * 16;
    int n0 = blockIdx.x * 64;
    v8f acc[4];
    gemm_block_bf16(W, Act, DI, 48, n0, lds, acc, m0);
    if (m0 >= 48) return;
    int mb = m0 + ((lane >> 4) ? 8 : 0);
#pragma unroll
    for (int nt = 0; nt < 4; ++nt) {
        int n = n0 + nt * 16 + (lane & 15);
        if (m0 == 0) {
            alignas(16) u16 pk[8];
#pragma unroll
            for (int r = 0; r < 8; ++r) pk[r] = f2bf(acc[nt][r]);
            *reinterpret_cast<uint4*>(dtpad + (size_t)n * 32 + mb) =
                *reinterpret_cast<const uint4*>(pk);
            // zero the K-padding half
            *reinterpret_cast<uint4*>(dtpad + (size_t)n * 32 + mb + 16) =
                make_uint4(0u, 0u, 0u, 0u);
        } else {
            float* dst = ((m0 == 16) ? Bm : Cm) + (size_t)n * DS + (mb - m0);
            *reinterpret_cast<float4*>(dst) =
                make_float4(acc[nt][0], acc[nt][1], acc[nt][2], acc[nt][3]);
            *reinterpret_cast<float4*>(dst + 4) =
                make_float4(acc[nt][4], acc[nt][5], acc[nt][6], acc[nt][7]);
        }
    }
}

// delta: softplus(dt @ dtproj^T + b) -> delta [dir][ch][t] f32  (K=32 padded)
__global__ __launch_bounds__(256) void k_gemm_delta(
    const u16* __restrict__ W, const u16* __restrict__ Act,
    const float* __restrict__ bias, float* __restrict__ delta)
{
    __shared__ alignas(16) u16 lds[GEMM_LDS_U16];
    int wave = threadIdx.x >> 5, lane = threadIdx.x & 31;
    int m0 = (blockIdx.y * 8 + wave) * 16;
    int n0 = blockIdx.x * 64;
    v8f acc[4];
    gemm_block_bf16(W, Act, 32, DI, n0, lds, acc, m0);
    int mb = m0 + ((lane >> 4) ? 8 : 0);
#pragma unroll
    for (int nt = 0; nt < 4; ++nt) {
        int n = n0 + nt * 16 + (lane & 15);
        int dir = n >> 12, t = n & 4095;
#pragma unroll
        for (int r = 0; r < 8; ++r) {
            float vv = acc[nt][r] + bias[mb + r];
            float sp = (vv > 20.f) ? vv : __logf(1.f + __expf(vv));
            delta[((size_t)dir * DI + mb + r) * L_SEQ + t] = sp;
        }
    }
}

// ===========================================================================
// Fused 3x3 + 1x1 conv stage (implicit GEMM, tap decomposition) -> cat[pix][256] bf16
// One block per output row h. 8 waves; wave = 16-row out-channel tile.
// Staging converts f32 -> bf16, so it uses the classic VGPR->LDS path.
// ===========================================================================
__global__ __launch_bounds__(256) void k_conv(
    const float* __restrict__ cur, const float* __restrict__ prev,
    const u16* __restrict__ w3p, const float* __restrict__ b3f,
    const u16* __restrict__ w1p, const float* __restrict__ b1f,
    u16* __restrict__ cat)
{
    __shared__ alignas(16) u16 xt[3 * 66 * 32];   // [row 3][col 66][ic 32]
    const int h = blockIdx.x;
    const int tid = threadIdx.x, wave = tid >> 5, lane = tid & 31;
    const int m0 = wave * 16;
    const int col = lane & 15, kb = (lane >> 4) * 16, koff = (lane >> 4) * 16;

    v8f z = {0.f, 0.f, 0.f, 0.f, 0.f, 0.f, 0.f, 0.f};
    v8f acc3[4] = {z, z, z, z};
    v8f acc1[4] = {z, z, z, z};

    for (int ic0 = 0; ic0 < 512; ic0 += 32) {
        if (tid < 198) {
            int r = tid / 66, c = tid % 66;
            int hh = h + r - 1, ww = c - 1;
            bool ok = (hh >= 0) & (hh < 64) & (ww >= 0) & (ww < 64);
            u16* dst = xt + (r * 66 + c) * 32;
#pragma unroll 8
            for (int j = 0; j < 32; ++j) {
                int ic = ic0 + j;
                float vv = 0.f;
                if (ok) {
                    const float* src = (ic < 256) ? (cur + (size_t)ic * 4096)
                                                  : (prev + (size_t)(ic - 256) * 4096);
                    vv = src[hh * 64 + ww];
                }
                dst[j] = f2bf(vv);
            }
        }
        __syncthreads();

#pragma unroll
        for (int tap = 0; tap < 9; ++tap) {
            const int kh = tap / 3, kw = tap % 3;
            FragAB a3;
            const u16* wp = w3p + ((size_t)tap * 128 + m0 + (lane & 15)) * 512 + ic0 + koff;
            a3.q[0] = *reinterpret_cast<const uint4*>(wp);
            a3.q[1] = *reinterpret_cast<const uint4*>(wp + 8);
            FragAB b[4];
#pragma unroll
            for (int nt = 0; nt < 4; ++nt) {
                const u16* bp = xt + ((kh * 66 + nt * 16 + col + kw) * 32) + kb;
                b[nt].q[0] = *reinterpret_cast<const uint4*>(bp);
                b[nt].q[1] = *reinterpret_cast<const uint4*>(bp + 8);
            }
#pragma unroll
            for (int nt = 0; nt < 4; ++nt)
                acc3[nt] = wmma_bf16(a3, b[nt], acc3[nt]);
            if (tap == 4) {
                FragAB a1;
                const u16* w1 = w1p + (size_t)(m0 + (lane & 15)) * 512 + ic0 + koff;
                a1.q[0] = *reinterpret_cast<const uint4*>(w1);
                a1.q[1] = *reinterpret_cast<const uint4*>(w1 + 8);
#pragma unroll
                for (int nt = 0; nt < 4; ++nt)
                    acc1[nt] = wmma_bf16(a1, b[nt], acc1[nt]);
            }
        }
        __syncthreads();
    }

    const int mb = m0 + ((lane >> 4) ? 8 : 0);
#pragma unroll
    for (int nt = 0; nt < 4; ++nt) {
        int w = nt * 16 + col;
        int p = h * 64 + w;
        alignas(16) u16 p3[8], p1[8];
#pragma unroll
        for (int r = 0; r < 8; ++r) {
            p3[r] = f2bf(fmaxf(acc3[nt][r] + b3f[mb + r], 0.f));
            p1[r] = f2bf(fmaxf(acc1[nt][r] + b1f[mb + r], 0.f));
        }
        *reinterpret_cast<uint4*>(cat + (size_t)p * 256 + mb) =
            *reinterpret_cast<const uint4*>(p3);
        *reinterpret_cast<uint4*>(cat + (size_t)p * 256 + 128 + mb) =
            *reinterpret_cast<const uint4*>(p1);
    }
}

// ===========================================================================
// LayerNorm over C=256: one wave per pixel. f32 in -> bf16 out.
// ===========================================================================
__global__ __launch_bounds__(256) void k_ln(
    const float* __restrict__ g2, const float* __restrict__ g,
    const float* __restrict__ b, u16* __restrict__ zln)
{
    int wave = threadIdx.x >> 5, lane = threadIdx.x & 31;
    int p = blockIdx.x * 8 + wave;
    const float* row = g2 + (size_t)p * 256;
    float4 v0 = *reinterpret_cast<const float4*>(row + lane * 8);
    float4 v1 = *reinterpret_cast<const float4*>(row + lane * 8 + 4);
    float vals[8] = {v0.x, v0.y, v0.z, v0.w, v1.x, v1.y, v1.z, v1.w};
    float s = 0.f, sq = 0.f;
#pragma unroll
    for (int j = 0; j < 8; ++j) { s += vals[j]; sq += vals[j] * vals[j]; }
#pragma unroll
    for (int off = 16; off >= 1; off >>= 1) {
        s  += __shfl_xor(s, off, 32);
        sq += __shfl_xor(sq, off, 32);
    }
    float mu  = s * (1.f / 256.f);
    float var = sq * (1.f / 256.f) - mu * mu;
    float inv = rsqrtf(var + EPS);
    alignas(16) u16 pk[8];
#pragma unroll
    for (int j = 0; j < 8; ++j) {
        int c = lane * 8 + j;
        pk[j] = f2bf((vals[j] - mu) * inv * g[c] + b[c]);
    }
    *reinterpret_cast<uint4*>(zln + (size_t)p * 256 + lane * 8) =
        *reinterpret_cast<const uint4*>(pk);
}

// ===========================================================================
// Build the 4 direction-permuted sequences (uint4 = 8 bf16 channels per thread)
// ===========================================================================
__global__ void k_permute(const u16* __restrict__ zln, u16* __restrict__ zp) {
    int idx = blockIdx.x * blockDim.x + threadIdx.x;  // over 16384*32
    if (idx >= NT_TOT * 32) return;
    int c8 = idx & 31;
    int nt = idx >> 5;
    int t = nt & 4095, dir = nt >> 12;
    int p;
    if (dir == 0)      p = t;
    else if (dir == 1) p = (t & 63) * 64 + (t >> 6);
    else if (dir == 2) p = 4095 - t;
    else { int u = 4095 - t; p = (u & 63) * 64 + (u >> 6); }
    *reinterpret_cast<uint4*>(zp + (size_t)nt * 256 + c8 * 8) =
        *reinterpret_cast<const uint4*>(zln + (size_t)p * 256 + c8 * 8);
}

// ===========================================================================
// Depthwise causal conv1d (K=4) + SiLU; repack to [dir*L][512] bf16
// ===========================================================================
__global__ void k_conv1d(const float* __restrict__ xi_raw, const float* __restrict__ cw,
                         const float* __restrict__ cb, u16* __restrict__ xi_conv) {
    size_t idx = (size_t)blockIdx.x * blockDim.x + threadIdx.x;  // 4*512*4096
    if (idx >= (size_t)NDIR * DI * L_SEQ) return;
    int t = idx & 4095;
    int ch = (idx >> 12) & 511;
    int dir = (int)(idx >> 21);
    const float* src = xi_raw + ((size_t)dir * DI + ch) * L_SEQ;
    float s = cb[ch];
#pragma unroll
    for (int j = 0; j < 4; ++j) {
        int tt = t - 3 + j;
        if (tt >= 0) s += cw[ch * 4 + j] * src[tt];
    }
    float y = s / (1.f + __expf(-s));
    xi_conv[((size_t)dir * L_SEQ + t) * DI + ch] = f2bf(y);
}

// ===========================================================================
// Selective scan: sequential over L. thread = (dir, channel); h[16] in regs.
// B/C staged in 64-step LDS chunks. Fuses  y = (scan + x*D) * silu(zg).
// ===========================================================================
__global__ __launch_bounds__(128) void k_scan(
    const float* __restrict__ delta, const u16* __restrict__ xi_conv,
    const float* __restrict__ Bm, const float* __restrict__ Cm,
    const float* __restrict__ Aneg, const float* __restrict__ Dp,
    const float* __restrict__ szg, u16* __restrict__ ymod)
{
    __shared__ float bs[64 * 16];
    __shared__ float cs[64 * 16];
    const int dir = blockIdx.x >> 2;
    const int d = (blockIdx.x & 3) * 128 + threadIdx.x;

    float A[16];
#pragma unroll
    for (int s = 0; s < 16; ++s) A[s] = Aneg[d * 16 + s];
    const float Dd = Dp[d];
    float h[16];
#pragma unroll
    for (int s = 0; s < 16; ++s) h[s] = 0.f;

    const float* dl = delta + ((size_t)dir * DI + d) * L_SEQ;

    for (int t0 = 0; t0 < L_SEQ; t0 += 64) {
        for (int i = threadIdx.x; i < 64 * 16; i += 128) {
            bs[i] = Bm[((size_t)dir * L_SEQ + t0) * DS + i];
            cs[i] = Cm[((size_t)dir * L_SEQ + t0) * DS + i];
        }
        __syncthreads();
        for (int tt = 0; tt < 64; ++tt) {
            const int t = t0 + tt;
            const float del = dl[t];
            const float x = bf2f(xi_conv[((size_t)dir * L_SEQ + t) * DI + d]);
            const float dx = del * x;
            float y = 0.f;
#pragma unroll
            for (int s = 0; s < 16; ++s) {
                float dA = __expf(del * A[s]);
                h[s] = dA * h[s] + dx * bs[tt * 16 + s];
                y += h[s] * cs[tt * 16 + s];
            }
            const float zv = szg[((size_t)dir * L_SEQ + t) * DI + d];
            ymod[((size_t)dir * L_SEQ + t) * DI + d] = f2bf((y + x * Dd) * zv);
        }
        __syncthreads();
    }
}

// ===========================================================================
// Final: inverse permutations, mean over 4 dirs, residual add with cur_bev.
// ===========================================================================
__global__ void k_final(const float* __restrict__ cur, const float* __restrict__ o2,
                        float* __restrict__ out) {
    int idx = blockIdx.x * blockDim.x + threadIdx.x;  // 256*4096
    if (idx >= C_CH * L_SEQ) return;
    int p = idx & 4095, c = idx >> 12;
    int tcol = (p & 63) * 64 + (p >> 6);
    int t0 = p;
    int t1 = tcol;
    int t2 = 4095 - p;
    int t3 = 4095 - tcol;
    float s = o2[(size_t)(0 * L_SEQ + t0) * 256 + c]
            + o2[(size_t)(1 * L_SEQ + t1) * 256 + c]
            + o2[(size_t)(2 * L_SEQ + t2) * 256 + c]
            + o2[(size_t)(3 * L_SEQ + t3) * 256 + c];
    out[idx] = cur[idx] + 0.25f * s;
}

// ===========================================================================
// Host launch
// ===========================================================================
extern "C" void kernel_launch(void* const* d_in, const int* in_sizes, int n_in,
                              void* d_out, int out_size, void* d_ws, size_t ws_size,
                              hipStream_t stream) {
    (void)in_sizes; (void)n_in; (void)out_size; (void)ws_size;
    const float* cur      = (const float*)d_in[0];
    const float* prev     = (const float*)d_in[1];
    const float* conv3_w  = (const float*)d_in[2];
    const float* bn3_g    = (const float*)d_in[3];
    const float* bn3_b    = (const float*)d_in[4];
    const float* bn3_m    = (const float*)d_in[5];
    const float* bn3_v    = (const float*)d_in[6];
    const float* conv1_w  = (const float*)d_in[7];
    const float* bn1_g    = (const float*)d_in[8];
    const float* bn1_b    = (const float*)d_in[9];
    const float* bn1_m    = (const float*)d_in[10];
    const float* bn1_v    = (const float*)d_in[11];
    const float* fuse_w   = (const float*)d_in[12];
    const float* bnf_g    = (const float*)d_in[13];
    const float* bnf_b    = (const float*)d_in[14];
    const float* bnf_m    = (const float*)d_in[15];
    const float* bnf_v    = (const float*)d_in[16];
    const float* seqp_w   = (const float*)d_in[17];
    const float* seqp_b   = (const float*)d_in[18];
    const float* ln_g     = (const float*)d_in[19];
    const float* ln_b     = (const float*)d_in[20];
    const float* inproj_w = (const float*)d_in[21];
    const float* conv1d_w = (const float*)d_in[22];
    const float* conv1d_b = (const float*)d_in[23];
    const float* xproj_w  = (const float*)d_in[24];
    const float* dtproj_w = (const float*)d_in[25];
    const float* dtproj_b = (const float*)d_in[26];
    const float* A_log    = (const float*)d_in[27];
    const float* Dp       = (const float*)d_in[28];
    const float* m_out_w  = (const float*)d_in[29];
    const float* outp_w   = (const float*)d_in[30];
    const float* outp_b   = (const float*)d_in[31];

    char* base = (char*)d_ws;
    size_t off = 0;
    auto alloc = [&](size_t bytes) -> void* {
        void* pp = base + off;
        off = (off + bytes + 255) & ~(size_t)255;
        return pp;
    };

    u16*   w3p    = (u16*)  alloc((size_t)9 * 128 * 512 * 2);
    float* b3f    = (float*)alloc(128 * 4);
    u16*   w1p    = (u16*)  alloc((size_t)128 * 512 * 2);
    float* b1f    = (float*)alloc(128 * 4);
    u16*   wfp    = (u16*)  alloc((size_t)256 * 256 * 2);
    float* bff    = (float*)alloc(256 * 4);
    u16*   seqpw  = (u16*)  alloc((size_t)256 * 256 * 2);
    u16*   inpp   = (u16*)  alloc((size_t)1024 * 256 * 2);
    u16*   xpp    = (u16*)  alloc((size_t)48 * 512 * 2);
    u16*   dtpp   = (u16*)  alloc((size_t)512 * 32 * 2);
    u16*   mop    = (u16*)  alloc((size_t)256 * 512 * 2);
    u16*   outpp  = (u16*)  alloc((size_t)256 * 256 * 2);
    float* Aneg   = (float*)alloc((size_t)512 * 16 * 4);
    u16*   cat    = (u16*)  alloc((size_t)L_SEQ * 256 * 2);
    u16*   y1     = (u16*)  alloc((size_t)L_SEQ * 256 * 2);
    float* g2out  = (float*)alloc((size_t)L_SEQ * 256 * 4);
    u16*   zln    = (u16*)  alloc((size_t)L_SEQ * 256 * 2);
    u16*   zp     = (u16*)  alloc((size_t)NT_TOT * 256 * 2);
    float* xi_raw = (float*)alloc((size_t)NDIR * DI * L_SEQ * 4);
    float* szg    = (float*)alloc((size_t)NT_TOT * DI * 4);
    u16*   xiconv = (u16*)  alloc((size_t)NT_TOT * DI * 2);
    u16*   dtpad  = (u16*)  alloc((size_t)NT_TOT * 32 * 2);
    float* BmBuf  = (float*)alloc((size_t)NT_TOT * DS * 4);
    float* CmBuf  = (float*)alloc((size_t)NT_TOT * DS * 4);
    float* delta  = (float*)alloc((size_t)NDIR * DI * L_SEQ * 4);
    u16*   ymod   = (u16*)  alloc((size_t)NT_TOT * DI * 2);
    u16*   o1     = (u16*)  alloc((size_t)NT_TOT * 256 * 2);
    float* o2     = (float*)alloc((size_t)NT_TOT * 256 * 4);

    // ---- weight prep ----
    k_cast_bf16<<<256, 256, 0, stream>>>(seqp_w,   seqpw, 256 * 256);
    k_cast_bf16<<<1024, 256, 0, stream>>>(inproj_w, inpp, 1024 * 256);
    k_cast_bf16<<<96, 256, 0, stream>>>(xproj_w,  xpp,   48 * 512);
    k_cast_bf16<<<512, 256, 0, stream>>>(m_out_w,  mop,  256 * 512);
    k_cast_bf16<<<256, 256, 0, stream>>>(outp_w,   outpp, 256 * 256);
    k_prep_c3<<<(9 * 128 * 512 + 255) / 256, 256, 0, stream>>>(
        conv3_w, bn3_g, bn3_b, bn3_m, bn3_v, w3p, b3f);
    k_prep_c1<<<(128 * 512 + 255) / 256, 256, 0, stream>>>(
        conv1_w, bn1_g, bn1_b, bn1_m, bn1_v, 128, 512, w1p, b1f);
    k_prep_c1<<<(256 * 256 + 255) / 256, 256, 0, stream>>>(
        fuse_w, bnf_g, bnf_b, bnf_m, bnf_v, 256, 256, wfp, bff);
    k_prep_dt<<<(512 * 32 + 255) / 256, 256, 0, stream>>>(dtproj_w, dtpp);
    k_prep_A<<<(512 * 16 + 255) / 256, 256, 0, stream>>>(A_log, Aneg);

    // ---- conv stage -> cat [4096][256] bf16 ----
    k_conv<<<64, 256, 0, stream>>>(cur, prev, w3p, b3f, w1p, b1f, cat);

    // ---- fuse 1x1 + bn + relu -> y1 bf16 ----
    k_gemm_tbf16<<<dim3(L_SEQ / 64, 2), 256, 0, stream>>>(wfp, cat, bff, y1, 256, 256, 1);

    // ---- seqp GEMM + bias -> g2out f32 ; then LayerNorm -> zln bf16 ----
    k_gemm_f32<<<dim3(L_SEQ / 64, 2), 256, 0, stream>>>(seqpw, y1, seqp_b, g2out, 256, 256);
    k_ln<<<L_SEQ / 8, 256, 0, stream>>>(g2out, ln_g, ln_b, zln);

    // ---- 4-direction permute -> zp ----
    k_permute<<<(NT_TOT * 32 + 255) / 256, 256, 0, stream>>>(zln, zp);

    // ---- in_proj GEMM (M=1024) ----
    k_gemm_inproj<<<dim3(NT_TOT / 64, 8), 256, 0, stream>>>(inpp, zp, xi_raw, szg);

    // ---- depthwise conv1d + SiLU ----
    k_conv1d<<<(int)(((size_t)NDIR * DI * L_SEQ + 255) / 256), 256, 0, stream>>>(
        xi_raw, conv1d_w, conv1d_b, xiconv);

    // ---- xproj GEMM (M=48) -> dt/B/C ----
    k_gemm_xproj<<<dim3(NT_TOT / 64, 1), 256, 0, stream>>>(xpp, xiconv, dtpad, BmBuf, CmBuf);

    // ---- delta = softplus(dt @ dtproj^T + b) ----
    k_gemm_delta<<<dim3(NT_TOT / 64, 4), 256, 0, stream>>>(dtpp, dtpad, dtproj_b, delta);

    // ---- selective scan (+ D skip + silu(zg) gate) -> ymod bf16 ----
    k_scan<<<16, 128, 0, stream>>>(delta, xiconv, BmBuf, CmBuf, Aneg, Dp, szg, ymod);

    // ---- m_out GEMM -> o1 bf16 ----
    k_gemm_tbf16<<<dim3(NT_TOT / 64, 2), 256, 0, stream>>>(mop, ymod, nullptr, o1, 512, 256, 0);

    // ---- outp GEMM + bias -> o2 f32 ----
    k_gemm_f32<<<dim3(NT_TOT / 64, 2), 256, 0, stream>>>(outpp, o1, outp_b, o2, 256, 256);

    // ---- inverse perms, mean over dirs, residual ----
    k_final<<<(C_CH * L_SEQ + 255) / 256, 256, 0, stream>>>(cur, o2, (float*)d_out);
}
